// DKM_73100343377958
// MI455X (gfx1250) — matrix operand order repaired and checked by reading, not compiled, not run on hardware
//
#include <hip/hip_runtime.h>
#include <hip/hip_bf16.h>
#include <math.h>

typedef __attribute__((ext_vector_type(16))) __bf16 v16bf;
typedef __attribute__((ext_vector_type(2)))  __bf16 v2bf;
typedef __attribute__((ext_vector_type(8)))  float  v8f;

#define EPS 1e-4f

// ---- hardware packed f32->bf16 conversion (1 VALU op per 2 elements) ----
__device__ __forceinline__ v2bf cvt2(float a, float b) {
#if __has_builtin(__builtin_amdgcn_cvt_pk_bf16_f32)
  return __builtin_amdgcn_cvt_pk_bf16_f32(a, b);
#else
  v2bf r; r[0] = (__bf16)a; r[1] = (__bf16)b; return r;
#endif
}
__device__ __forceinline__ unsigned cvt2_bits(float a, float b) {
  union { v2bf p; unsigned u; } o; o.p = cvt2(a, b); return o.u;
}
// pack 8 floats (two float4) into elements [base, base+8) of a v16bf
__device__ __forceinline__ void pack8(v16bf& v, int base, float4 a, float4 b) {
  union { v16bf v; v2bf p[8]; } u; u.v = v;
  u.p[(base >> 1) + 0] = cvt2(a.x, a.y);
  u.p[(base >> 1) + 1] = cvt2(a.z, a.w);
  u.p[(base >> 1) + 2] = cvt2(b.x, b.y);
  u.p[(base >> 1) + 3] = cvt2(b.z, b.w);
  v = u.v;
}

// ---------------------------------------------------------------------------
// Init: per-row squared norms of W
// ---------------------------------------------------------------------------
__global__ void dkm_wsq_kernel(const float* __restrict__ W,
                               float* __restrict__ wsq, int N) {
  int i = blockIdx.x * blockDim.x + threadIdx.x;
  if (i >= N) return;
  const float4* p = (const float4*)(W + (size_t)i * 64);
  float s = 0.f;
#pragma unroll
  for (int j = 0; j < 16; ++j) {
    float4 v = p[j];
    s += v.x * v.x + v.y * v.y + v.z * v.z + v.w * v.w;
  }
  wsq[i] = s;
}

// ---- shared helper: refresh csq + bf16 cluster tables from Cws (tid<64) ----
__device__ __forceinline__ void refresh_cluster_tables(
    int tid, const float* Cws, float* csq,
    unsigned short* Cbf, unsigned short* CbfT) {
  if (tid < 64) {
    float s = 0.f;
#pragma unroll
    for (int j = 0; j < 16; ++j) {
      float4 v = *(const float4*)(Cws + tid * 64 + j * 4);
      s += v.x * v.x + v.y * v.y + v.z * v.z + v.w * v.w;
      unsigned lo = cvt2_bits(v.x, v.y);
      unsigned hi = cvt2_bits(v.z, v.w);
      *(unsigned*)&Cbf[tid * 64 + j * 4 + 0] = lo;
      *(unsigned*)&Cbf[tid * 64 + j * 4 + 2] = hi;
      CbfT[(j * 4 + 0) * 64 + tid] = (unsigned short)lo;
      CbfT[(j * 4 + 1) * 64 + tid] = (unsigned short)(lo >> 16);
      CbfT[(j * 4 + 2) * 64 + tid] = (unsigned short)hi;
      CbfT[(j * 4 + 3) * 64 + tid] = (unsigned short)(hi >> 16);
    }
    csq[tid] = s;
  }
}

// ---------------------------------------------------------------------------
// Init: copy clusters, build csq + bf16 tables, zero accumulators, clear flag
// ---------------------------------------------------------------------------
__global__ __launch_bounds__(256) void dkm_init_kernel(
    const float* __restrict__ C0, float* __restrict__ Cws,
    float* __restrict__ csq, float* __restrict__ acc,
    float* __restrict__ colsum, int* __restrict__ flags,
    unsigned short* __restrict__ Cbf, unsigned short* __restrict__ CbfT) {
  const int tid = threadIdx.x;
#pragma unroll
  for (int it = 0; it < 16; ++it) {
    int i = tid + it * 256;
    Cws[i] = C0[i];
    acc[i] = 0.f;
  }
  __syncthreads();
  refresh_cluster_tables(tid, Cws, csq, Cbf, CbfT);
  if (tid < 64) colsum[tid] = 0.f;
  if (tid == 0) flags[0] = 0;
}

// ---------------------------------------------------------------------------
// Fused per-iteration kernel:
//   phase 1: G = W_tile @ C^T (bf16 WMMA) -> d2 -> softmax -> attn
//   phase 2: block-partial W^T @ attn (bf16 WMMA, output [d][k]) -> atomics
// 256 threads = 8 waves; wave handles 16 rows; block handles 128 rows.
// ---------------------------------------------------------------------------
__global__ __launch_bounds__(256) void dkm_iter_kernel(
    const float* __restrict__ W, const float* __restrict__ wsq,
    const unsigned short* __restrict__ Cbf, const float* __restrict__ csq,
    float* __restrict__ attn_out, float* __restrict__ acc,
    float* __restrict__ colsum, const int* __restrict__ flags, int N) {
  if (flags[0]) return;  // converged on an earlier iteration: freeze state

  __shared__ unsigned short attnT[64 * 136];  // bf16 bits, [k][n_local], padded
  __shared__ unsigned short Wt[64 * 136];     // bf16 bits, [d][n_local], padded

  const int lane = threadIdx.x & 31;
  const int wave = threadIdx.x >> 5;
  const int lm   = lane & 15;
  const int half = lane >> 4;
  const int rowBlock = blockIdx.x * 128;
  const int waveRow  = rowBlock + wave * 16;

  // ---- load A fragments (16 rows x 64 d of W, bf16); transpose into LDS ----
  v16bf afrag[2];
  {
    const int arow = waveRow + lm;
    const int n0 = wave * 16 + lm;
#pragma unroll
    for (int ds = 0; ds < 2; ++ds) {
      const float* p = W + (size_t)arow * 64 + ds * 32 + half * 8;
#pragma unroll
      for (int g = 0; g < 2; ++g) {
        float4 x0 = *(const float4*)(p + g * 16);
        float4 x1 = *(const float4*)(p + g * 16 + 4);
        pack8(afrag[ds], g * 8, x0, x1);
      }
      // scatter this lane's 16 (d, value) pairs into W^T LDS tile
      union { v16bf v; unsigned short us[16]; } uw;
      uw.v = afrag[ds];
#pragma unroll
      for (int e = 0; e < 16; ++e) {
        const int d = ds * 32 + (e >> 3) * 16 + half * 8 + (e & 7);
        Wt[d * 136 + n0] = uw.us[e];
      }
    }
  }

  // ---- phase 1 WMMAs: G[n][k], 4 k-tiles x 2 K-steps ----
  v8f gacc[4];
#pragma unroll
  for (int kt = 0; kt < 4; ++kt) {
    v8f c = {};
#pragma unroll
    for (int ds = 0; ds < 2; ++ds) {
      // B element j: column k = kt*16+lm, K-index d = ds*32 + (j>>3)*16 + half*8 + (j&7)
      union { v16bf v; uint4 q[2]; } ub;
      const unsigned short* q = Cbf + (kt * 16 + lm) * 64 + ds * 32 + half * 8;
      ub.q[0] = *(const uint4*)(q + 0);
      ub.q[1] = *(const uint4*)(q + 16);
      c = __builtin_amdgcn_wmma_f32_16x16x32_bf16(false, afrag[ds], false, ub.v,
                                                  (short)0, c, false, false);
    }
    gacc[kt] = c;
  }

  // ---- d2 -> dist -> softmax over k (row lives in 16 lanes x 4 kt) ----
  float sqc[4];
#pragma unroll
  for (int kt = 0; kt < 4; ++kt) sqc[kt] = csq[kt * 16 + lm];

  float attnv[4][8];
#pragma unroll
  for (int r = 0; r < 8; ++r) {
    const float wq = wsq[waveRow + half * 8 + r];
    float sv[4];
    float m = -3.0e38f;
#pragma unroll
    for (int kt = 0; kt < 4; ++kt) {
      float d2 = wq + sqc[kt] - 2.0f * gacc[kt][r];
      d2 = fmaxf(d2, 0.0f);
      sv[kt] = -sqrtf(d2);  // TEMPERATURE == 1
      m = fmaxf(m, sv[kt]);
    }
#pragma unroll
    for (int off = 1; off < 16; off <<= 1) m = fmaxf(m, __shfl_xor(m, off, 32));
    float sum = 0.f;
#pragma unroll
    for (int kt = 0; kt < 4; ++kt) {
      float e = __expf(sv[kt] - m);
      sv[kt] = e;
      sum += e;
    }
#pragma unroll
    for (int off = 1; off < 16; off <<= 1) sum += __shfl_xor(sum, off, 32);
    float inv = 1.0f / sum;
#pragma unroll
    for (int kt = 0; kt < 4; ++kt) attnv[kt][r] = sv[kt] * inv;
  }

  // ---- write attn (d_out region), colsum atomics, attn^T tile to LDS ----
#pragma unroll
  for (int kt = 0; kt < 4; ++kt) {
    const int k = kt * 16 + lm;
    float ps = 0.f;
#pragma unroll
    for (int r = 0; r < 8; ++r) {
      attn_out[(size_t)(waveRow + half * 8 + r) * 64 + k] = attnv[kt][r];
      ps += attnv[kt][r];
    }
    ps += __shfl_xor(ps, 16, 32);
    if (half == 0) atomicAdd(&colsum[k], ps);
    uint4 pk;
    pk.x = cvt2_bits(attnv[kt][0], attnv[kt][1]);
    pk.y = cvt2_bits(attnv[kt][2], attnv[kt][3]);
    pk.z = cvt2_bits(attnv[kt][4], attnv[kt][5]);
    pk.w = cvt2_bits(attnv[kt][6], attnv[kt][7]);
    *(uint4*)&attnT[k * 136 + wave * 16 + half * 8] = pk;
  }
  __syncthreads();

  // ---- phase 2: block-partial W^T @ attn -> acc[k*64+d] (M=d, N=k, K=n) ----
  // All fragments are contiguous 16B LDS vector loads now.
#pragma unroll
  for (int t = 0; t < 2; ++t) {
    const int tile = wave * 2 + t;
    const int dt = tile >> 2;
    const int kt = tile & 3;
    const unsigned short* arow = &Wt[(dt * 16 + lm) * 136];
    const unsigned short* brow = &attnT[(kt * 16 + lm) * 136];
    v8f c2 = {};
#pragma unroll
    for (int nc = 0; nc < 4; ++nc) {
      union { v16bf v; uint4 q[2]; } ua;  // A = W^T: K = g*16 + half*8 + within
      ua.q[0] = *(const uint4*)(arow + nc * 32 + half * 8);
      ua.q[1] = *(const uint4*)(arow + nc * 32 + 16 + half * 8);
      union { v16bf v; uint4 q[2]; } ub;  // B = attn: K = g*16 + half*8 + within
      ub.q[0] = *(const uint4*)(brow + nc * 32 + half * 8);
      ub.q[1] = *(const uint4*)(brow + nc * 32 + 16 + half * 8);
      c2 = __builtin_amdgcn_wmma_f32_16x16x32_bf16(false, ua.v, false, ub.v,
                                                   (short)0, c2, false, false);
    }
    const int k = kt * 16 + lm;
#pragma unroll
    for (int r = 0; r < 8; ++r) {
      const int d = dt * 16 + half * 8 + r;
      atomicAdd(&acc[k * 64 + d], c2[r]);
    }
  }
}

// ---------------------------------------------------------------------------
// Per-iteration cluster update + convergence + re-zero accumulators
// ---------------------------------------------------------------------------
__global__ __launch_bounds__(256) void dkm_update_kernel(
    float* __restrict__ Cws, float* __restrict__ csq, float* __restrict__ acc,
    float* __restrict__ colsum, int* __restrict__ flags,
    unsigned short* __restrict__ Cbf, unsigned short* __restrict__ CbfT) {
  if (flags[0]) return;
  __shared__ float red[256];
  const int tid = threadIdx.x;
  float newc[16];
  float local = 0.f;
#pragma unroll
  for (int it = 0; it < 16; ++it) {
    int i = tid + it * 256;
    float nc = acc[i] / colsum[i >> 6];
    float dd = Cws[i] - nc;
    local += dd * dd;
    newc[it] = nc;
  }
  red[tid] = local;
  __syncthreads();
  for (int s = 128; s > 0; s >>= 1) {
    if (tid < s) red[tid] += red[tid + s];
    __syncthreads();
  }
  const bool conv = sqrtf(red[0]) <= EPS;
  if (!conv) {
#pragma unroll
    for (int it = 0; it < 16; ++it) Cws[tid + it * 256] = newc[it];
  } else if (tid == 0) {
    flags[0] = 1;  // break BEFORE assigning new clusters (matches reference)
  }
  __syncthreads();
  refresh_cluster_tables(tid, Cws, csq, Cbf, CbfT);
  if (tid < 64) colsum[tid] = 0.f;
#pragma unroll
  for (int it = 0; it < 16; ++it) acc[tid + it * 256] = 0.f;
}

// ---------------------------------------------------------------------------
// Final: compressed = attn @ clusters (bf16 WMMA), B from transposed bf16 table
// ---------------------------------------------------------------------------
__global__ __launch_bounds__(256) void dkm_compress_kernel(
    const float* __restrict__ attn, const unsigned short* __restrict__ CbfT,
    float* __restrict__ comp, int N) {
  const int lane = threadIdx.x & 31;
  const int wave = threadIdx.x >> 5;
  const int lm = lane & 15, half = lane >> 4;
  const int waveRow = blockIdx.x * 128 + wave * 16;

  v16bf afrag[2];
  {
    const int arow = waveRow + lm;
#pragma unroll
    for (int ds = 0; ds < 2; ++ds) {
      const float* p = attn + (size_t)arow * 64 + ds * 32 + half * 8;
#pragma unroll
      for (int g = 0; g < 2; ++g) {
        float4 x0 = *(const float4*)(p + g * 16);
        float4 x1 = *(const float4*)(p + g * 16 + 4);
        pack8(afrag[ds], g * 8, x0, x1);
      }
    }
  }
#pragma unroll
  for (int dt = 0; dt < 4; ++dt) {
    const int d = dt * 16 + lm;
    v8f c = {};
#pragma unroll
    for (int ds = 0; ds < 2; ++ds) {
      // B element j: K-index kc = ds*32 + (j>>3)*16 + half*8 + (j&7), column d
      union { v16bf v; uint4 q[2]; } ub;
      const unsigned short* q = CbfT + d * 64 + ds * 32 + half * 8;
      ub.q[0] = *(const uint4*)(q + 0);
      ub.q[1] = *(const uint4*)(q + 16);
      c = __builtin_amdgcn_wmma_f32_16x16x32_bf16(false, afrag[ds], false, ub.v,
                                                  (short)0, c, false, false);
    }
#pragma unroll
    for (int r = 0; r < 8; ++r)
      comp[(size_t)(waveRow + half * 8 + r) * 64 + d] = c[r];
  }
}

__global__ void dkm_copy_clusters_kernel(const float* __restrict__ Cws,
                                         float* __restrict__ out) {
  int i = threadIdx.x;
#pragma unroll
  for (int it = 0; it < 16; ++it) out[i + it * 256] = Cws[i + it * 256];
}

// ---------------------------------------------------------------------------
extern "C" void kernel_launch(void* const* d_in, const int* in_sizes, int n_in,
                              void* d_out, int out_size, void* d_ws, size_t ws_size,
                              hipStream_t stream) {
  const float* W  = (const float*)d_in[0];   // [N,64] fp32
  const float* C0 = (const float*)d_in[1];   // [64,64] fp32
  const int N = in_sizes[0] / 64;

  float* out = (float*)d_out;
  float* comp      = out;                    // [N*64]
  float* clust_out = out + (size_t)N * 64;   // [64*64]
  float* attn_out  = clust_out + 64 * 64;    // [N*64] — written live each iter

  float* ws     = (float*)d_ws;
  float* Cws    = ws;                        // [4096] current clusters (f32)
  float* csq    = ws + 4096;                 // [64]
  float* colsum = ws + 4096 + 64;            // [64]
  float* acc    = ws + 4096 + 128;           // [4096] W^T @ attn accumulator
  float* wsq    = ws + 4096 + 128 + 4096;    // [N]
  int*   flags  = (int*)(wsq + N);           // [1] done flag
  unsigned short* Cbf  = (unsigned short*)(flags + 16);  // [64*64] bf16 clusters
  unsigned short* CbfT = Cbf + 64 * 64;                  // [64*64] bf16 clusters^T

  dkm_wsq_kernel<<<(N + 255) / 256, 256, 0, stream>>>(W, wsq, N);
  dkm_init_kernel<<<1, 256, 0, stream>>>(C0, Cws, csq, acc, colsum, flags, Cbf, CbfT);
  for (int it = 0; it < 10; ++it) {
    dkm_iter_kernel<<<N / 128, 256, 0, stream>>>(W, wsq, Cbf, csq, attn_out,
                                                 acc, colsum, flags, N);
    dkm_update_kernel<<<1, 256, 0, stream>>>(Cws, csq, acc, colsum, flags, Cbf, CbfT);
  }
  dkm_compress_kernel<<<N / 128, 256, 0, stream>>>(attn_out, CbfT, comp, N);
  dkm_copy_clusters_kernel<<<1, 256, 0, stream>>>(Cws, clust_out);
}
